// DeformableSamplingModule_40450001994333
// MI455X (gfx1250) — matrix operand order repaired and checked by reading, not compiled, not run on hardware
//
#include <hip/hip_runtime.h>
#include <hip/hip_bf16.h>
#include <stdint.h>

#define N_Q     100000
#define C_CH    256
#define D_IN    258
#define D_PAD   288
#define HID_N   256
#define P_PTS   8
#define H_IMG   512
#define W_IMG   512
#define TILE_M  16
#define RADIUS_F 2.0f

// LDS row strides (in dwords). 145 % 64 and 129 % 64 are odd -> conflict-free
// across 16 rows. Rows hold packed bf16 pairs.
#define XROW_DW 145   // 290 bf16 slots, holds 288 (padded D) or 256 (h2)
#define HROW_DW 129   // 258 bf16 slots, holds 256 (h1)

typedef __attribute__((ext_vector_type(16))) __bf16 v16bf;
typedef __attribute__((ext_vector_type(8)))  float  v8f;

union Frag16 { v16bf v; uint32_t u[8]; };

__device__ __forceinline__ uint16_t f2bf(float f) {
  union { float f; uint32_t u; } c; c.f = f;
  return (uint16_t)((c.u + 0x7FFFu + ((c.u >> 16) & 1u)) >> 16);
}

// A fragment (16x32 bf16, M x K) from an LDS tile stored row-major as packed
// bf16 pairs. Per ISA: lane(0-15)=row M, kg=lane>>4 selects K sub-block.
//  elements 0-7 : K = ks*32 + kg*8 + e
//  elements 8-15: K = ks*32 + 16 + kg*8 + (e-8)
__device__ __forceinline__ void loadA(Frag16& A, const uint32_t* rowp, int ks, int kg) {
  int base = ks * 16 + kg * 4;
#pragma unroll
  for (int d = 0; d < 4; ++d) {
    A.u[d]     = rowp[base + d];
    A.u[d + 4] = rowp[base + 8 + d];
  }
}

// B fragment (32x16 bf16, K x N) from the pre-swizzled weight buffer:
// frag = fragIdx, per lane 16 consecutive bf16 (32B) -> two b128 loads.
__device__ __forceinline__ void loadB(Frag16& B, const uint16_t* frags, int fragIdx, int lane) {
  const uint4* p = (const uint4*)(frags + (((size_t)fragIdx * 32 + lane) * 16));
  uint4 lo = p[0], hi = p[1];
  B.u[0] = lo.x; B.u[1] = lo.y; B.u[2] = lo.z; B.u[3] = lo.w;
  B.u[4] = hi.x; B.u[5] = hi.y; B.u[6] = hi.z; B.u[7] = hi.w;
}

__device__ __forceinline__ v8f wmma_bf16(const Frag16& A, const Frag16& B, v8f acc) {
  return __builtin_amdgcn_wmma_f32_16x16x32_bf16(false, A.v, false, B.v,
                                                 (short)0, acc, false, false);
}

// ---------------------------------------------------------------------------
// Weight pre-swizzle: fp32 (K x Ncols, row-major) -> bf16 WMMA-B fragments.
// dst linear layout == (ntile*ksteps + ks)*512 + lane*16 + e,
// with element e of lane l mapping to (K = ks*32 + (e&8)*2 + (l>>4)*8 + (e&7),
//                                      N = ntile*16 + (l&15)).
// Out-of-range K/N (padding) written as zero.
// ---------------------------------------------------------------------------
__global__ void swizzle_weights(const float* __restrict__ src, uint16_t* __restrict__ dst,
                                int Krows, int Ncols, int ksteps, int ntiles) {
  int tid = blockIdx.x * blockDim.x + threadIdx.x;
  int total = ntiles * ksteps * 512;
  if (tid >= total) return;
  int e    = tid & 15;
  int lane = (tid >> 4) & 31;
  int frag = tid >> 9;
  int ntile = frag / ksteps;
  int ks    = frag - ntile * ksteps;
  int n = ntile * 16 + (lane & 15);
  int k = ks * 32 + (e & 8) * 2 + (lane >> 4) * 8 + (e & 7);
  float v = (k < Krows && n < Ncols) ? src[(size_t)k * Ncols + n] : 0.0f;
  dst[tid] = f2bf(v);
}

// ---------------------------------------------------------------------------
// Main fused kernel: one 256-thread block (8 wave32) per 16-query tile.
// ---------------------------------------------------------------------------
__global__ __launch_bounds__(256)
void deform_main(const float* __restrict__ gq,      // (N, 256)
                 const float* __restrict__ proj,    // (N, 2)
                 const float* __restrict__ fmap,    // (H, W, 256)
                 const float* __restrict__ ln_g0, const float* __restrict__ ln_b0,
                 const float* __restrict__ b1_0,  const float* __restrict__ b2_0,
                 const float* __restrict__ b3_0,   // 16
                 const float* __restrict__ ln_g1, const float* __restrict__ ln_b1v,
                 const float* __restrict__ b1_1,  const float* __restrict__ b2_1,
                 const float* __restrict__ b3_1,   // 8
                 const uint16_t* __restrict__ w1s0, const uint16_t* __restrict__ w2s0,
                 const uint16_t* __restrict__ w3s0,
                 const uint16_t* __restrict__ w1s1, const uint16_t* __restrict__ w2s1,
                 const uint16_t* __restrict__ w3s1,
                 float* __restrict__ out) {
  __shared__ uint32_t sA[2 * 16 * XROW_DW];  // x_ln (stage1 in), then h2 (stage3 in)
  __shared__ uint32_t sB[2 * 16 * HROW_DW];  // h1
  __shared__ float    soff[16 * 16];         // 16 queries x (P*2) offsets
  __shared__ float    swl [16 * 8];          // weight-head logits
  __shared__ float    snw [16 * 8];          // normalized softmax weights

  const int tid  = threadIdx.x;
  const int wave = tid >> 5;
  const int lane = tid & 31;
  const int kg   = lane >> 4;       // k-group for WMMA A/B lane mapping
  const int lrow = lane & 15;
  const int qbase = blockIdx.x * TILE_M;

  // ---------------- Stage 0: LayerNorm (both affine sets) -> sA (bf16) ------
  {
    int r = wave * 2 + kg;          // lanes 0-15 row wave*2, lanes 16-31 row +1
    int q = qbase + r;
    float xv[18];
    float s = 0.f, ss = 0.f;
    int cnt = 0;
    for (int e = lrow; e < D_IN; e += 16) {
      float x = (e < C_CH) ? gq[(size_t)q * C_CH + e] : proj[(size_t)q * 2 + (e - C_CH)];
      xv[cnt++] = x;
      s += x; ss += x * x;
    }
#pragma unroll
    for (int m = 8; m >= 1; m >>= 1) {   // reduce within each 16-lane half
      s  += __shfl_xor(s,  m, 32);
      ss += __shfl_xor(ss, m, 32);
    }
    const float inv_d = 1.0f / (float)D_IN;
    float mean = s * inv_d;
    float var  = ss * inv_d - mean * mean;
    float rstd = rsqrtf(var + 1e-5f);

    uint16_t* sA16 = (uint16_t*)sA;
    int i = 0;
    for (int e = lrow; e < D_PAD; e += 16, ++i) {
      uint16_t v0 = 0, v1 = 0;
      if (e < D_IN) {
        float xn = (xv[i] - mean) * rstd;
        v0 = f2bf(xn * ln_g0[e] + ln_b0[e]);
        v1 = f2bf(xn * ln_g1[e] + ln_b1v[e]);
      }
      sA16[(size_t)(0 * 16 + r) * (2 * XROW_DW) + e] = v0;
      sA16[(size_t)(1 * 16 + r) * (2 * XROW_DW) + e] = v1;
    }
  }
  __syncthreads();

  // ---------------- Stage 1: h1 = relu(x_ln @ W1 + b1)  (K=288, 9 ksteps) ---
  {
    int mlp = wave >> 2;       // 0 = offsets MLP, 1 = weights MLP
    int ww  = wave & 3;        // 4 waves per MLP, 64 cols each
    const uint16_t* w1 = mlp ? w1s1 : w1s0;
    const float*    b1 = mlp ? b1_1 : b1_0;
    const uint32_t* rowp = sA + (size_t)(mlp * 16 + lrow) * XROW_DW;

    v8f acc[4];
#pragma unroll
    for (int t = 0; t < 4; ++t) acc[t] = 0.f;

    for (int ks = 0; ks < 9; ++ks) {
      Frag16 A; loadA(A, rowp, ks, kg);
#pragma unroll
      for (int t = 0; t < 4; ++t) {
        Frag16 B; loadB(B, w1, (ww * 4 + t) * 9 + ks, lane);
        acc[t] = wmma_bf16(A, B, acc[t]);
      }
    }

    uint16_t* sB16 = (uint16_t*)sB;
#pragma unroll
    for (int t = 0; t < 4; ++t) {
      int col = (ww * 4 + t) * 16 + lrow;
      float bias = b1[col];
#pragma unroll
      for (int v = 0; v < 8; ++v) {
        int row = v + kg * 8;
        float x = acc[t][v] + bias;
        x = x > 0.f ? x : 0.f;
        sB16[(size_t)(mlp * 16 + row) * (2 * HROW_DW) + col] = f2bf(x);
      }
    }
  }
  __syncthreads();

  // ---------------- Stage 2: h2 = relu(h1 @ W2 + b2)  (K=256, 8 ksteps) -----
  {
    int mlp = wave >> 2;
    int ww  = wave & 3;
    const uint16_t* w2 = mlp ? w2s1 : w2s0;
    const float*    b2 = mlp ? b2_1 : b2_0;
    const uint32_t* rowp = sB + (size_t)(mlp * 16 + lrow) * HROW_DW;

    v8f acc[4];
#pragma unroll
    for (int t = 0; t < 4; ++t) acc[t] = 0.f;

    for (int ks = 0; ks < 8; ++ks) {
      Frag16 A; loadA(A, rowp, ks, kg);
#pragma unroll
      for (int t = 0; t < 4; ++t) {
        Frag16 B; loadB(B, w2, (ww * 4 + t) * 8 + ks, lane);
        acc[t] = wmma_bf16(A, B, acc[t]);
      }
    }

    uint16_t* sA16 = (uint16_t*)sA;   // overwrite x_ln with h2
#pragma unroll
    for (int t = 0; t < 4; ++t) {
      int col = (ww * 4 + t) * 16 + lrow;
      float bias = b2[col];
#pragma unroll
      for (int v = 0; v < 8; ++v) {
        int row = v + kg * 8;
        float x = acc[t][v] + bias;
        x = x > 0.f ? x : 0.f;
        sA16[(size_t)(mlp * 16 + row) * (2 * XROW_DW) + col] = f2bf(x);
      }
    }
  }
  __syncthreads();

  // ---------------- Stage 3: heads (wave 0: offsets 16 cols, wave 4: wgts) --
  if ((wave & 3) == 0) {
    int mlp = wave >> 2;
    const uint16_t* w3 = mlp ? w3s1 : w3s0;
    const uint32_t* rowp = sA + (size_t)(mlp * 16 + lrow) * XROW_DW;

    v8f acc = 0.f;
    for (int ks = 0; ks < 8; ++ks) {
      Frag16 A; loadA(A, rowp, ks, kg);
      Frag16 B; loadB(B, w3, ks, lane);
      acc = wmma_bf16(A, B, acc);
    }

    int col = lrow;
    if (mlp == 0) {
      float bias = b3_0[col];
#pragma unroll
      for (int v = 0; v < 8; ++v) {
        int row = v + kg * 8;
        soff[row * 16 + col] = tanhf(acc[v] + bias) * RADIUS_F;
      }
    } else if (col < P_PTS) {
      float bias = b3_1[col];
#pragma unroll
      for (int v = 0; v < 8; ++v) {
        int row = v + kg * 8;
        swl[row * 8 + col] = acc[v] + bias;
      }
    }
  }
  __syncthreads();

  // ---------------- softmax over P per query --------------------------------
  if (tid < 16) {
    float mx = -3.0e38f;
#pragma unroll
    for (int p = 0; p < P_PTS; ++p) mx = fmaxf(mx, swl[tid * 8 + p]);
    float ex[P_PTS];
    float sum = 0.f;
#pragma unroll
    for (int p = 0; p < P_PTS; ++p) { ex[p] = expf(swl[tid * 8 + p] - mx); sum += ex[p]; }
    float inv = 1.0f / fmaxf(sum, 1e-8f);
#pragma unroll
    for (int p = 0; p < P_PTS; ++p) snw[tid * 8 + p] = ex[p] * inv;
  }
  __syncthreads();

  // ---------------- Stage 4: bilinear gather + weighted reduce --------------
  {
    const int c0 = lane * 8;     // each lane owns 8 consecutive channels
    for (int qi = 0; qi < 2; ++qi) {
      int r = wave * 2 + qi;
      int q = qbase + r;
      float px = proj[(size_t)q * 2 + 0];
      float py = proj[(size_t)q * 2 + 1];
      float4 a0 = make_float4(0.f, 0.f, 0.f, 0.f);
      float4 a1 = make_float4(0.f, 0.f, 0.f, 0.f);

      for (int p = 0; p < P_PTS; ++p) {
        float cx = px + soff[r * 16 + p * 2 + 0];
        float cy = py + soff[r * 16 + p * 2 + 1];
        float gx = fminf(fmaxf(2.0f * cx / (float)(W_IMG - 1) - 1.0f, -1.1f), 1.1f);
        float gy = fminf(fmaxf(2.0f * cy / (float)(H_IMG - 1) - 1.0f, -1.1f), 1.1f);
        float ix = fminf(fmaxf((gx + 1.0f) * 0.5f * (float)(W_IMG - 1), 0.0f), (float)(W_IMG - 1));
        float iy = fminf(fmaxf((gy + 1.0f) * 0.5f * (float)(H_IMG - 1), 0.0f), (float)(H_IMG - 1));
        float x0 = floorf(ix), y0 = floorf(iy);
        float wx = ix - x0,   wy = iy - y0;
        int x0i = (int)x0, y0i = (int)y0;
        int x1i = min(x0i + 1, W_IMG - 1);
        int y1i = min(y0i + 1, H_IMG - 1);
        float nwp = snw[r * 8 + p];
        float c00 = nwp * (1.f - wx) * (1.f - wy);
        float c01 = nwp * wx         * (1.f - wy);
        float c10 = nwp * (1.f - wx) * wy;
        float c11 = nwp * wx         * wy;

        const float4* p00 = (const float4*)(fmap + ((size_t)(y0i * W_IMG + x0i)) * C_CH + c0);
        const float4* p01 = (const float4*)(fmap + ((size_t)(y0i * W_IMG + x1i)) * C_CH + c0);
        const float4* p10 = (const float4*)(fmap + ((size_t)(y1i * W_IMG + x0i)) * C_CH + c0);
        const float4* p11 = (const float4*)(fmap + ((size_t)(y1i * W_IMG + x1i)) * C_CH + c0);
        float4 v00a = p00[0], v00b = p00[1];
        float4 v01a = p01[0], v01b = p01[1];
        float4 v10a = p10[0], v10b = p10[1];
        float4 v11a = p11[0], v11b = p11[1];

        a0.x += c00 * v00a.x + c01 * v01a.x + c10 * v10a.x + c11 * v11a.x;
        a0.y += c00 * v00a.y + c01 * v01a.y + c10 * v10a.y + c11 * v11a.y;
        a0.z += c00 * v00a.z + c01 * v01a.z + c10 * v10a.z + c11 * v11a.z;
        a0.w += c00 * v00a.w + c01 * v01a.w + c10 * v10a.w + c11 * v11a.w;
        a1.x += c00 * v00b.x + c01 * v01b.x + c10 * v10b.x + c11 * v11b.x;
        a1.y += c00 * v00b.y + c01 * v01b.y + c10 * v10b.y + c11 * v11b.y;
        a1.z += c00 * v00b.z + c01 * v01b.z + c10 * v10b.z + c11 * v11b.z;
        a1.w += c00 * v00b.w + c01 * v01b.w + c10 * v10b.w + c11 * v11b.w;
      }

      float4* o = (float4*)(out + (size_t)q * C_CH + c0);
      o[0] = a0;
      o[1] = a1;
    }
  }
}

// ---------------------------------------------------------------------------
// Host-side launch
// ---------------------------------------------------------------------------
extern "C" void kernel_launch(void* const* d_in, const int* in_sizes, int n_in,
                              void* d_out, int out_size, void* d_ws, size_t ws_size,
                              hipStream_t stream) {
  const float* gq       = (const float*)d_in[0];
  const float* proj     = (const float*)d_in[1];
  const float* fmap     = (const float*)d_in[2];
  const float* off_ln_g = (const float*)d_in[3];
  const float* off_ln_b = (const float*)d_in[4];
  const float* off_W1   = (const float*)d_in[5];
  const float* off_b1   = (const float*)d_in[6];
  const float* off_W2   = (const float*)d_in[7];
  const float* off_b2   = (const float*)d_in[8];
  const float* off_W3   = (const float*)d_in[9];
  const float* off_b3   = (const float*)d_in[10];
  const float* wgt_ln_g = (const float*)d_in[11];
  const float* wgt_ln_b = (const float*)d_in[12];
  const float* wgt_W1   = (const float*)d_in[13];
  const float* wgt_b1   = (const float*)d_in[14];
  const float* wgt_W2   = (const float*)d_in[15];
  const float* wgt_b2   = (const float*)d_in[16];
  const float* wgt_W3   = (const float*)d_in[17];
  const float* wgt_b3   = (const float*)d_in[18];

  // Workspace layout (bf16 element counts):
  //   W1 frags: 16 ntiles * 9 ksteps * 512 = 73728
  //   W2 frags: 16 ntiles * 8 ksteps * 512 = 65536
  //   W3 frags:  1 ntile  * 8 ksteps * 512 =  4096
  uint16_t* ws = (uint16_t*)d_ws;
  const size_t W1SZ = 73728, W2SZ = 65536, W3SZ = 4096;
  uint16_t* w1s0 = ws;
  uint16_t* w2s0 = w1s0 + W1SZ;
  uint16_t* w3s0 = w2s0 + W2SZ;
  uint16_t* w1s1 = w3s0 + W3SZ;
  uint16_t* w2s1 = w1s1 + W1SZ;
  uint16_t* w3s1 = w2s1 + W2SZ;

  // Pre-swizzle weights -> bf16 WMMA-B fragments (runs every call; cheap).
  swizzle_weights<<<(int)(W1SZ / 256), 256, 0, stream>>>(off_W1, w1s0, D_IN, HID_N, 9, 16);
  swizzle_weights<<<(int)(W2SZ / 256), 256, 0, stream>>>(off_W2, w2s0, HID_N, HID_N, 8, 16);
  swizzle_weights<<<(int)(W3SZ / 256), 256, 0, stream>>>(off_W3, w3s0, HID_N, 2 * P_PTS, 8, 1);
  swizzle_weights<<<(int)(W1SZ / 256), 256, 0, stream>>>(wgt_W1, w1s1, D_IN, HID_N, 9, 16);
  swizzle_weights<<<(int)(W2SZ / 256), 256, 0, stream>>>(wgt_W2, w2s1, HID_N, HID_N, 8, 16);
  swizzle_weights<<<(int)(W3SZ / 256), 256, 0, stream>>>(wgt_W3, w3s1, HID_N, P_PTS, 8, 1);

  // Main fused kernel: 100000 / 16 = 6250 tiles exactly.
  deform_main<<<N_Q / TILE_M, 256, 0, stream>>>(
      gq, proj, fmap,
      off_ln_g, off_ln_b, off_b1, off_b2, off_b3,
      wgt_ln_g, wgt_ln_b, wgt_b1, wgt_b2, wgt_b3,
      w1s0, w2s0, w3s0, w1s1, w2s1, w3s1,
      (float*)d_out);
}
